// TransformerBlock_81235011436686
// MI455X (gfx1250) — matrix operand order repaired and checked by reading, not compiled, not run on hardware
//
#include <hip/hip_runtime.h>
#include <hip/hip_bf16.h>

typedef _Float16 v16h  __attribute__((ext_vector_type(16)));
typedef _Float16 f16x8 __attribute__((ext_vector_type(8)));
typedef float    v8f   __attribute__((ext_vector_type(8)));
typedef int      v4i   __attribute__((ext_vector_type(4)));

#define Bb    4
#define Ss    4096
#define LOG_S 12
#define Dd    1024
#define LOG_D 10
#define Hh    16
#define Ee    8
#define HID   4096
#define Tt    (Bb * Ss)   // 16384

#define AS1 __attribute__((address_space(1)))
#define AS3 __attribute__((address_space(3)))

#if __has_builtin(__builtin_amdgcn_global_load_async_to_lds_b128) && \
    __has_builtin(__builtin_amdgcn_s_wait_asynccnt)
#define HAVE_ASYNC 1
#endif

// 16-byte global(f16) -> LDS copy; async (ASYNCcnt-tracked) on CDNA5
static __device__ __forceinline__ void cp16_g2l(const _Float16* g, _Float16* l) {
#ifdef HAVE_ASYNC
  _Float16* gg = const_cast<_Float16*>(g);
  __builtin_amdgcn_global_load_async_to_lds_b128((AS1 v4i*)gg, (AS3 v4i*)l, 0, 0);
#else
  *(f16x8*)l = *(const f16x8*)g;
#endif
}

static __device__ __forceinline__ void wait_async_lds() {
#ifdef HAVE_ASYNC
  __builtin_amdgcn_s_wait_asynccnt(0);
#endif
}

// ---------------------------------------------------------------------------
// Weight conversion kernels (fp32 -> f16, with layout permutation)
// ---------------------------------------------------------------------------
__global__ void cvt_copy_kernel(const float* __restrict__ src,
                                _Float16* __restrict__ dst, int n) {
  int i = blockIdx.x * blockDim.x + threadIdx.x;
  if (i < n) dst[i] = (_Float16)src[i];
}

// src: (E, K, N) fp32   ->  dst: (E, N, K) f16
__global__ void cvt_transpose_kernel(const float* __restrict__ src,
                                     _Float16* __restrict__ dst,
                                     int N, int K, int total) {
  int i = blockIdx.x * blockDim.x + threadIdx.x;
  if (i >= total) return;
  int nk = N * K;
  int e = i / nk, r = i - e * nk;
  int n = r / K, k = r - n * K;
  dst[i] = (_Float16)src[(size_t)e * nk + (size_t)k * N + n];
}

// conv_w (D_out, D_in, 3) fp32 -> convT (D_out, 3*D_in) f16, k-major blocks
__global__ void cvt_conv_kernel(const float* __restrict__ src,
                                _Float16* __restrict__ dst) {
  int i = blockIdx.x * blockDim.x + threadIdx.x;
  int total = Dd * 3 * Dd;
  if (i >= total) return;
  int o = i / (3 * Dd), r = i - o * 3 * Dd;
  int kk = r >> LOG_D, ci = r & (Dd - 1);
  dst[i] = (_Float16)src[((size_t)o * Dd + ci) * 3 + kk];
}

// ---------------------------------------------------------------------------
// Generic WMMA GEMM:  Out[M,N] = act(A[M,K] @ B^T + bias)
//   B is f16 (N,K) row-major (per-expert stride bStride)
//   block tile 128x128, K-tile 32, 256 threads = 8 waves (4 M x 2 N),
//   double-buffered LDS pipeline, async global->LDS B copies (CDNA5),
//   each wave: 2x4 tiles of v_wmma_f32_16x16x32_f16
// ---------------------------------------------------------------------------
template <bool A_F16, bool GATHER, bool CONV, bool RELU, bool OUT_F16>
__global__ __launch_bounds__(256) void wmma_gemm_kernel(
    const void* __restrict__ Asrc, const _Float16* __restrict__ Bsrc,
    const float* __restrict__ bias, void* __restrict__ Out,
    int M, int N, int K, int ldA,
    const int* __restrict__ cntPtr, const int* __restrict__ offPtr,
    const int* __restrict__ gatherIdx,
    long long bStride, int biasStride,
    const float* __restrict__ convX) {
  __shared__ _Float16 sA[2 * 128 * 32];
  __shared__ _Float16 sB[2 * 128 * 32];

  const int e = blockIdx.z;
  const int mBase = blockIdx.x * 128;
  const int nBase = blockIdx.y * 128;
  const int Mv = cntPtr ? cntPtr[e] : M;
  if (mBase >= Mv) return;
  const int rowOff = offPtr ? offPtr[e] : 0;
  const _Float16* Bp = Bsrc + (size_t)e * (size_t)bStride;
  const float* bp = bias ? bias + (size_t)e * biasStride : nullptr;

  const int tid = threadIdx.x;
  const int lr = tid >> 1;        // 0..127 : tile row (A) / tile col (B)
  const int lh = tid & 1;         // which 16-wide K chunk
  const int lane = tid & 31;
  const int wave = tid >> 5;
  const int wm = (wave & 3) * 32; // wave row block within 128
  const int wn = (wave >> 2) * 64;// wave col block within 128

  const int cr = mBase + lr;      // compact row this thread loads for A
  const bool rvalid = (cr < Mv);
  long long arow;
  if (GATHER) arow = rvalid ? (long long)gatherIdx[rowOff + cr] : 0;
  else        arow = (long long)rowOff + cr;

  const int nk = K / 32;

  // stage tile kt into LDS buffer `buf`
  auto load_tile = [&](int kt, int buf) {
    const int k0 = kt * 32 + lh * 16;
    _Float16* sAb = sA + buf * (128 * 32);
    _Float16* sBb = sB + buf * (128 * 32);
    // ------------- A tile (fp32 -> f16 convert, or f16 async copy) -------
    if (CONV) {
      // powers of two: strength-reduced index math (S=4096, D=1024)
      const int kk = k0 >> LOG_D;
      const int i0 = k0 & (Dd - 1);
      const int bb = cr >> LOG_S, ss = cr & (Ss - 1);
      const int srow = ss + kk - 1;
      float4 f0, f1, f2, f3;
      if (rvalid && srow >= 0 && srow < Ss) {
        const float4* src =
            (const float4*)(convX + (((size_t)(bb << LOG_S) + srow) << LOG_D) + i0);
        f0 = src[0]; f1 = src[1]; f2 = src[2]; f3 = src[3];
      } else {
        f0 = f1 = f2 = f3 = make_float4(0.f, 0.f, 0.f, 0.f);
      }
      _Float16* d = sAb + lr * 32 + lh * 16;
      d[0]=(_Float16)f0.x; d[1]=(_Float16)f0.y; d[2]=(_Float16)f0.z; d[3]=(_Float16)f0.w;
      d[4]=(_Float16)f1.x; d[5]=(_Float16)f1.y; d[6]=(_Float16)f1.z; d[7]=(_Float16)f1.w;
      d[8]=(_Float16)f2.x; d[9]=(_Float16)f2.y; d[10]=(_Float16)f2.z; d[11]=(_Float16)f2.w;
      d[12]=(_Float16)f3.x; d[13]=(_Float16)f3.y; d[14]=(_Float16)f3.z; d[15]=(_Float16)f3.w;
    } else if (A_F16) {
      const _Float16* src = (const _Float16*)Asrc + arow * (long long)ldA + k0;
      _Float16* d = sAb + lr * 32 + lh * 16;
#ifdef HAVE_ASYNC
      if (rvalid) {   // invalid rows keep stale LDS; their results are masked
        cp16_g2l(src, d);
        cp16_g2l(src + 8, d + 8);
      }
#else
      f16x8 z = (f16x8)(_Float16)0.0f;
      f16x8 u0 = z, u1 = z;
      if (rvalid) { u0 = ((const f16x8*)src)[0]; u1 = ((const f16x8*)src)[1]; }
      *(f16x8*)d = u0;
      *(f16x8*)(d + 8) = u1;
#endif
    } else {
      float4 f0, f1, f2, f3;
      if (rvalid) {
        const float4* src =
            (const float4*)((const float*)Asrc + arow * (long long)ldA + k0);
        f0 = src[0]; f1 = src[1]; f2 = src[2]; f3 = src[3];
      } else {
        f0 = f1 = f2 = f3 = make_float4(0.f, 0.f, 0.f, 0.f);
      }
      _Float16* d = sAb + lr * 32 + lh * 16;
      d[0]=(_Float16)f0.x; d[1]=(_Float16)f0.y; d[2]=(_Float16)f0.z; d[3]=(_Float16)f0.w;
      d[4]=(_Float16)f1.x; d[5]=(_Float16)f1.y; d[6]=(_Float16)f1.z; d[7]=(_Float16)f1.w;
      d[8]=(_Float16)f2.x; d[9]=(_Float16)f2.y; d[10]=(_Float16)f2.z; d[11]=(_Float16)f2.w;
      d[12]=(_Float16)f3.x; d[13]=(_Float16)f3.y; d[14]=(_Float16)f3.z; d[15]=(_Float16)f3.w;
    }
    // ------------- B tile: async global -> LDS (f16, (N,K)-major) --------
    {
      const _Float16* src = Bp + (size_t)(nBase + lr) * K + k0;
      _Float16* d = sBb + lr * 32 + lh * 16;
      cp16_g2l(src, d);
      cp16_g2l(src + 8, d + 8);
#ifndef HAVE_ASYNC
      if (kt + 1 < nk)  // CDNA5 global_prefetch_b8 of next B K-slice
        __builtin_prefetch(src + 32, 0, 0);
#endif
    }
  };

  v8f acc[2][4] = {};
  load_tile(0, 0);

  for (int kt = 0; kt < nk; ++kt) {
    const int buf = kt & 1;
    wait_async_lds();     // our async copies into buf are done
    __syncthreads();      // everyone's copies into buf are done
    if (kt + 1 < nk) load_tile(kt + 1, buf ^ 1);  // other buffer: no barrier

    // ---------------- fragments (ISA 7.12.2 layouts) ----------------------
    const _Float16* sAb = sA + buf * (128 * 32);
    const _Float16* sBb = sB + buf * (128 * 32);
    const int am = lane & 15;      // M (A) / N (B) within 16-tile
    const int ah = lane >> 4;      // lane half -> K sub-select
    v16h afrag[2], bfrag[4];
#pragma unroll
    for (int mi = 0; mi < 2; ++mi) {
      const _Float16* p = sAb + (wm + mi * 16 + am) * 32 + ah * 8;
      f16x8 lo = *(const f16x8*)p;        // VGPRs 0..3 : K pairs
      f16x8 hi = *(const f16x8*)(p + 16); // VGPRs 4..7 : K+16 pairs
#pragma unroll
      for (int q = 0; q < 8; ++q) { afrag[mi][q] = lo[q]; afrag[mi][8 + q] = hi[q]; }
    }
#pragma unroll
    for (int ni = 0; ni < 4; ++ni) {
      const _Float16* p = sBb + (wn + ni * 16 + am) * 32 + ah * 8;
      f16x8 lo = *(const f16x8*)p;        // K pairs
      f16x8 hi = *(const f16x8*)(p + 16); // K+16 pairs
#pragma unroll
      for (int q = 0; q < 8; ++q) { bfrag[ni][q] = lo[q]; bfrag[ni][8 + q] = hi[q]; }
    }
#pragma unroll
    for (int mi = 0; mi < 2; ++mi)
#pragma unroll
      for (int ni = 0; ni < 4; ++ni)
        acc[mi][ni] = __builtin_amdgcn_wmma_f32_16x16x32_f16(
            false, afrag[mi], false, bfrag[ni], (short)0, acc[mi][ni], false,
            false);
  }

  // ---------------- epilogue: bias (+ReLU) and store ----------------------
  const int am = lane & 15;
  const int ah = lane >> 4;
#pragma unroll
  for (int mi = 0; mi < 2; ++mi) {
#pragma unroll
    for (int ni = 0; ni < 4; ++ni) {
      const int col = nBase + wn + ni * 16 + am;
      const float bv = bp ? bp[col] : 0.0f;
#pragma unroll
      for (int r = 0; r < 8; ++r) {
        const int row = mBase + wm + mi * 16 + ah * 8 + r;
        if (row < Mv) {
          float v = acc[mi][ni][r] + bv;
          if (RELU) v = fmaxf(v, 0.0f);
          const size_t off = (size_t)(rowOff + row) * N + col;
          if (OUT_F16) ((_Float16*)Out)[off] = (_Float16)v;
          else         ((float*)Out)[off] = v;
        }
      }
    }
  }
}

// ---------------------------------------------------------------------------
// Attention over the degenerate seq axis (length B=4). One wave per (n, h).
// ---------------------------------------------------------------------------
__device__ inline float wave_sum(float v) {
#pragma unroll
  for (int s = 16; s > 0; s >>= 1) v += __shfl_xor(v, s, 32);
  return v;
}

__global__ __launch_bounds__(32) void attn_kernel(const float* __restrict__ qkv,
                                                  float* __restrict__ o) {
  const int idx = blockIdx.x;        // S * H blocks
  const int n = idx & (Ss - 1);
  const int h = idx >> LOG_S;
  const int lane = threadIdx.x;      // 32 lanes, 2 dims each (hd = 64)
  float q[4][2], k[4][2], v[4][2];
#pragma unroll
  for (int l = 0; l < 4; ++l) {
    const size_t base = ((size_t)(l * Ss + n)) * (3 * Dd) + h * 64;
#pragma unroll
    for (int j = 0; j < 2; ++j) {
      const int d = lane + j * 32;
      q[l][j] = qkv[base + d];
      k[l][j] = qkv[base + Dd + d];
      v[l][j] = qkv[base + 2 * Dd + d];
    }
  }
  float sc[4][4];
#pragma unroll
  for (int l = 0; l < 4; ++l)
#pragma unroll
    for (int m = 0; m < 4; ++m)
      sc[l][m] = wave_sum(q[l][0] * k[m][0] + q[l][1] * k[m][1]) * 0.125f;
  float p[4][4];
#pragma unroll
  for (int l = 0; l < 4; ++l) {
    float mx = fmaxf(fmaxf(sc[l][0], sc[l][1]), fmaxf(sc[l][2], sc[l][3]));
    float s = 0.f;
#pragma unroll
    for (int m = 0; m < 4; ++m) { p[l][m] = __expf(sc[l][m] - mx); s += p[l][m]; }
    const float inv = 1.0f / s;
#pragma unroll
    for (int m = 0; m < 4; ++m) p[l][m] *= inv;
  }
#pragma unroll
  for (int l = 0; l < 4; ++l)
#pragma unroll
    for (int j = 0; j < 2; ++j) {
      float a = 0.f;
#pragma unroll
      for (int m = 0; m < 4; ++m) a += p[l][m] * v[m][j];
      o[((size_t)(l * Ss + n)) * Dd + h * 64 + lane + j * 32] = a;
    }
}

// ---------------------------------------------------------------------------
// residual add + LayerNorm (one block per token row)
// ---------------------------------------------------------------------------
__global__ __launch_bounds__(256) void add_ln_kernel(
    const float* __restrict__ a, const float* __restrict__ b,
    const float* __restrict__ g, const float* __restrict__ be,
    float* __restrict__ out) {
  __shared__ float red[256];
  const size_t row = blockIdx.x;
  const int tid = threadIdx.x;
  float v[4];
  float s = 0.f;
#pragma unroll
  for (int j = 0; j < 4; ++j) {
    const int i = tid + j * 256;
    v[j] = a[row * Dd + i] + b[row * Dd + i];
    s += v[j];
  }
  red[tid] = s; __syncthreads();
  for (int st = 128; st > 0; st >>= 1) { if (tid < st) red[tid] += red[tid + st]; __syncthreads(); }
  const float mean = red[0] * (1.0f / Dd);
  __syncthreads();
  float sq = 0.f;
#pragma unroll
  for (int j = 0; j < 4; ++j) { const float d = v[j] - mean; sq += d * d; }
  red[tid] = sq; __syncthreads();
  for (int st = 128; st > 0; st >>= 1) { if (tid < st) red[tid] += red[tid + st]; __syncthreads(); }
  const float rstd = rsqrtf(red[0] * (1.0f / Dd) + 1e-5f);
#pragma unroll
  for (int j = 0; j < 4; ++j) {
    const int i = tid + j * 256;
    out[row * Dd + i] = (v[j] - mean) * rstd * g[i] + be[i];
  }
}

// ---------------------------------------------------------------------------
// MoE gate: softmax(x @ gate_w + gate_b), top-2, renormalize, atomic ranks.
// One wave per token, 8 tokens per 256-thread block.
// ---------------------------------------------------------------------------
__global__ __launch_bounds__(256) void gate_kernel(
    const float* __restrict__ x, const float* __restrict__ gw,
    const float* __restrict__ gb, int* __restrict__ cnt,
    int* __restrict__ tokExp, int* __restrict__ tokRank,
    float* __restrict__ tokW) {
  const int wave = threadIdx.x >> 5, lane = threadIdx.x & 31;
  const int t = blockIdx.x * 8 + wave;
  if (t >= Tt) return;
  float acc[Ee] = {};
  for (int j = 0; j < 32; ++j) {
    const int d = j * 32 + lane;
    const float xv = x[(size_t)t * Dd + d];
#pragma unroll
    for (int e = 0; e < Ee; ++e) acc[e] += xv * gw[d * Ee + e];
  }
#pragma unroll
  for (int e = 0; e < Ee; ++e) acc[e] = wave_sum(acc[e]);
  if (lane == 0) {
    float logit[Ee], mx = -1e30f;
#pragma unroll
    for (int e = 0; e < Ee; ++e) { logit[e] = acc[e] + gb[e]; mx = fmaxf(mx, logit[e]); }
    float s = 0.f, p[Ee];
#pragma unroll
    for (int e = 0; e < Ee; ++e) { p[e] = __expf(logit[e] - mx); s += p[e]; }
    const float inv = 1.0f / s;
#pragma unroll
    for (int e = 0; e < Ee; ++e) p[e] *= inv;
    int e0 = 0; float p0 = p[0];
#pragma unroll
    for (int e = 1; e < Ee; ++e) if (p[e] > p0) { p0 = p[e]; e0 = e; }
    int e1 = (e0 == 0) ? 1 : 0; float p1 = p[e1];
#pragma unroll
    for (int e = 0; e < Ee; ++e)
      if (e != e0 && p[e] > p1) { p1 = p[e]; e1 = e; }
    const float w0 = p0 / (p0 + p1);
    tokExp[2 * t] = e0; tokExp[2 * t + 1] = e1;
    tokW[2 * t] = w0;   tokW[2 * t + 1] = 1.0f - w0;
    tokRank[2 * t] = atomicAdd(&cnt[e0], 1);
    tokRank[2 * t + 1] = atomicAdd(&cnt[e1], 1);
  }
}

__global__ void scan_kernel(const int* __restrict__ cnt, int* __restrict__ off) {
  off[0] = 0;
  for (int e = 0; e < Ee; ++e) off[e + 1] = off[e] + cnt[e];
}

__global__ __launch_bounds__(256) void slot_kernel(
    const int* __restrict__ tokExp, const int* __restrict__ tokRank,
    const int* __restrict__ off, int* __restrict__ tokSlot,
    int* __restrict__ tokOfSlot) {
  const int t = blockIdx.x * blockDim.x + threadIdx.x;
  if (t >= Tt) return;
#pragma unroll
  for (int i = 0; i < 2; ++i) {
    const int e = tokExp[2 * t + i];
    const int slot = off[e] + tokRank[2 * t + i];
    tokSlot[2 * t + i] = slot;
    tokOfSlot[slot] = t;
  }
}

// ---------------------------------------------------------------------------
// Combine top-2 expert outputs, residual, final LayerNorm -> d_out
// ---------------------------------------------------------------------------
__global__ __launch_bounds__(256) void combine_ln_kernel(
    const float* __restrict__ x1, const float* __restrict__ slotOut,
    const int* __restrict__ tokSlot, const float* __restrict__ tokW,
    const float* __restrict__ g, const float* __restrict__ be,
    float* __restrict__ out) {
  __shared__ float red[256];
  const size_t t = blockIdx.x;
  const int tid = threadIdx.x;
  const size_t s0 = (size_t)tokSlot[2 * t], s1 = (size_t)tokSlot[2 * t + 1];
  const float w0 = tokW[2 * t], w1 = tokW[2 * t + 1];
  float v[4];
  float s = 0.f;
#pragma unroll
  for (int j = 0; j < 4; ++j) {
    const int i = tid + j * 256;
    v[j] = x1[t * Dd + i] + w0 * slotOut[s0 * Dd + i] + w1 * slotOut[s1 * Dd + i];
    s += v[j];
  }
  red[tid] = s; __syncthreads();
  for (int st = 128; st > 0; st >>= 1) { if (tid < st) red[tid] += red[tid + st]; __syncthreads(); }
  const float mean = red[0] * (1.0f / Dd);
  __syncthreads();
  float sq = 0.f;
#pragma unroll
  for (int j = 0; j < 4; ++j) { const float d = v[j] - mean; sq += d * d; }
  red[tid] = sq; __syncthreads();
  for (int st = 128; st > 0; st >>= 1) { if (tid < st) red[tid] += red[tid + st]; __syncthreads(); }
  const float rstd = rsqrtf(red[0] * (1.0f / Dd) + 1e-5f);
#pragma unroll
  for (int j = 0; j < 4; ++j) {
    const int i = tid + j * 256;
    out[t * Dd + i] = (v[j] - mean) * rstd * g[i] + be[i];
  }
}

// ---------------------------------------------------------------------------
// Host launcher
// ---------------------------------------------------------------------------
static inline size_t align256(size_t x) { return (x + 255) & ~(size_t)255; }

extern "C" void kernel_launch(void* const* d_in, const int* in_sizes, int n_in,
                              void* d_out, int out_size, void* d_ws,
                              size_t ws_size, hipStream_t stream) {
  const float* x         = (const float*)d_in[0];
  const float* conv_w    = (const float*)d_in[1];
  const float* conv_b    = (const float*)d_in[2];
  const float* in_proj_w = (const float*)d_in[3];
  const float* in_proj_b = (const float*)d_in[4];
  const float* out_proj_w= (const float*)d_in[5];
  const float* out_proj_b= (const float*)d_in[6];
  const float* ln1_g     = (const float*)d_in[7];
  const float* ln1_b     = (const float*)d_in[8];
  const float* gate_w    = (const float*)d_in[9];
  const float* gate_b    = (const float*)d_in[10];
  const float* w1        = (const float*)d_in[11];
  const float* b1        = (const float*)d_in[12];
  const float* w2        = (const float*)d_in[13];
  const float* b2        = (const float*)d_in[14];
  const float* ln2_g     = (const float*)d_in[15];
  const float* ln2_b     = (const float*)d_in[16];
  float* out = (float*)d_out;

  // ---- workspace layout ----
  char* p = (char*)d_ws;
  size_t o = 0;
  auto alloc = [&](size_t bytes) { void* r = p + o; o = align256(o + bytes); return r; };
  _Float16* convT    = (_Float16*)alloc((size_t)Dd * 3 * Dd * 2);
  _Float16* inprojT  = (_Float16*)alloc((size_t)3 * Dd * Dd * 2);
  _Float16* outprojT = (_Float16*)alloc((size_t)Dd * Dd * 2);
  _Float16* w1t      = (_Float16*)alloc((size_t)Ee * HID * Dd * 2);
  _Float16* w2t      = (_Float16*)alloc((size_t)Ee * Dd * HID * 2);
  float*    xconv    = (float*)alloc((size_t)Tt * Dd * 4);
  float*    qkv      = (float*)alloc((size_t)Tt * 3 * Dd * 4);
  float*    attno    = (float*)alloc((size_t)Tt * Dd * 4);
  float*    attnp    = (float*)alloc((size_t)Tt * Dd * 4);
  float*    x1       = (float*)alloc((size_t)Tt * Dd * 4);
  _Float16* hidden   = (_Float16*)alloc((size_t)2 * Tt * HID * 2);
  float*    slotOut  = (float*)alloc((size_t)2 * Tt * Dd * 4);
  int*      cnt      = (int*)alloc(Ee * 4);
  int*      off      = (int*)alloc((Ee + 1) * 4);
  int*      tokExp   = (int*)alloc((size_t)2 * Tt * 4);
  int*      tokRank  = (int*)alloc((size_t)2 * Tt * 4);
  int*      tokSlot  = (int*)alloc((size_t)2 * Tt * 4);
  float*    tokW     = (float*)alloc((size_t)2 * Tt * 4);
  int*      tokOfSlot= (int*)alloc((size_t)2 * Tt * 4);
  (void)ws_size; (void)n_in; (void)in_sizes; (void)out_size;

  (void)hipMemsetAsync(cnt, 0, Ee * sizeof(int), stream);

  // ---- one-time fp32 -> f16 weight conversion / transposition ----
  cvt_copy_kernel<<<(3 * Dd * Dd + 255) / 256, 256, 0, stream>>>(in_proj_w, inprojT, 3 * Dd * Dd);
  cvt_copy_kernel<<<(Dd * Dd + 255) / 256, 256, 0, stream>>>(out_proj_w, outprojT, Dd * Dd);
  cvt_transpose_kernel<<<(Ee * HID * Dd + 255) / 256, 256, 0, stream>>>(w1, w1t, HID, Dd, Ee * HID * Dd);
  cvt_transpose_kernel<<<(Ee * Dd * HID + 255) / 256, 256, 0, stream>>>(w2, w2t, Dd, HID, Ee * Dd * HID);
  cvt_conv_kernel<<<(Dd * 3 * Dd + 255) / 256, 256, 0, stream>>>(conv_w, convT);

  // ---- conv1d as shifted GEMM: (T,1024) = X_pad(T,3072) @ convT^T ----
  wmma_gemm_kernel<false, false, true, false, false>
      <<<dim3(Tt / 128, Dd / 128, 1), 256, 0, stream>>>(
          x, convT, conv_b, xconv, Tt, Dd, 3 * Dd, 0,
          nullptr, nullptr, nullptr, 0, 0, x);

  // ---- QKV projection ----
  wmma_gemm_kernel<false, false, false, false, false>
      <<<dim3(Tt / 128, 3 * Dd / 128, 1), 256, 0, stream>>>(
          xconv, inprojT, in_proj_b, qkv, Tt, 3 * Dd, Dd, Dd,
          nullptr, nullptr, nullptr, 0, 0, nullptr);

  // ---- attention over length-4 axis ----
  attn_kernel<<<Ss * Hh, 32, 0, stream>>>(qkv, attno);

  // ---- output projection ----
  wmma_gemm_kernel<false, false, false, false, false>
      <<<dim3(Tt / 128, Dd / 128, 1), 256, 0, stream>>>(
          attno, outprojT, out_proj_b, attnp, Tt, Dd, Dd, Dd,
          nullptr, nullptr, nullptr, 0, 0, nullptr);

  // ---- residual + LN1 ----
  add_ln_kernel<<<Tt, 256, 0, stream>>>(xconv, attnp, ln1_g, ln1_b, x1);

  // ---- gate / top-2 routing ----
  gate_kernel<<<Tt / 8, 256, 0, stream>>>(x1, gate_w, gate_b, cnt, tokExp, tokRank, tokW);
  scan_kernel<<<1, 1, 0, stream>>>(cnt, off);
  slot_kernel<<<Tt / 256, 256, 0, stream>>>(tokExp, tokRank, off, tokSlot, tokOfSlot);

  // ---- expert FFN1: hidden = relu(gather(x1) @ w1[e]) , f16 out ----
  wmma_gemm_kernel<false, true, false, true, true>
      <<<dim3(Tt / 128, HID / 128, Ee), 256, 0, stream>>>(
          x1, w1t, b1, hidden, Tt, HID, Dd, Dd,
          cnt, off, tokOfSlot, (long long)HID * Dd, HID, nullptr);

  // ---- expert FFN2: slotOut = hidden @ w2[e] + b2 ----
  wmma_gemm_kernel<true, false, false, false, false>
      <<<dim3(Tt / 128, Dd / 128, Ee), 256, 0, stream>>>(
          hidden, w2t, b2, slotOut, Tt, Dd, HID, HID,
          cnt, off, nullptr, (long long)Dd * HID, Dd, nullptr);

  // ---- combine top-2 + residual + LN2 -> output ----
  combine_ln_kernel<<<Tt, 256, 0, stream>>>(x1, slotOut, tokSlot, tokW,
                                            ln2_g, ln2_b, out);
}
